// GPTSoftmax_86517821216451
// MI455X (gfx1250) — matrix operand order repaired and checked by reading, not compiled, not run on hardware
//
#include <hip/hip_runtime.h>
#include <hip/hip_bf16.h>
#include <math.h>

typedef _Float16 f16;
typedef _Float16 v16h __attribute__((ext_vector_type(16)));
typedef float    v8f  __attribute__((ext_vector_type(8)));
typedef unsigned int u32;

// Model constants (GPT-2 small, B=4, T=1024)
#define Bn   4
#define Tn   1024
#define Dn   768
#define Hn   12
#define DHn  64
#define Ln   12
#define Vn   50257
#define FFn  3072
#define Mn   (Bn*Tn)        // 4096 rows
#define VPAD 50304          // V padded to multiple of 64

union AF { v16h v; u32 u[8]; f16 h[16]; };

static __device__ inline v8f wmma_f16(const AF& a, const AF& b, v8f c) {
    // D = A(16x32 f16) * B(32x16 f16) + C(16x16 f32)
    return __builtin_amdgcn_wmma_f32_16x16x32_f16(
        false, a.v, false, b.v, (short)0, c, false, false);
}

// A-fragment K offsets for 16-bit A 16x32 (contiguous order, matches B packing):
// halves 0..15 -> K = kg*16 + {0..15}
static __device__ inline void make_koff(int kg, int* koff) {
#pragma unroll
    for (int vg = 0; vg < 8; ++vg)
        koff[vg] = 2 * vg + kg * 16;
}

// Async copy 16 bytes global -> LDS (gfx1250 VGLOBAL async op, ASYNCcnt).
static __device__ inline void async_cp_b128(u32 lds_byte_off, const f16* gaddr) {
    asm volatile("global_load_async_to_lds_b128 %0, %1, off"
                 :: "v"(lds_byte_off), "v"(gaddr) : "memory");
}
// Low 32 bits of a generic pointer into the LDS aperture = wave LDS byte offset.
static __device__ inline u32 lds_off32(const void* p) {
    return (u32)(size_t)p;
}

// ---------------------------------------------------------------------------
// Pack fp32 row-major W[K,N] into WMMA-B-fragment-native f16 tiles:
// out[((nt*Kt + kt)*32 + lane)*16 + h] = W[kt*32 + (lane/16)*16 + h][nt*16 + lane%16]
// ---------------------------------------------------------------------------
__global__ void gpt_pack_w(const float* __restrict__ W, f16* __restrict__ out,
                           int N, int Ktiles, int Ntiles) {
    size_t total = (size_t)Ntiles * Ktiles * 512;
    size_t i = (size_t)blockIdx.x * blockDim.x + threadIdx.x;
    if (i >= total) return;
    int hh   = (int)(i & 15);
    int lane = (int)((i >> 4) & 31);
    size_t tile = i >> 9;
    int kt = (int)(tile % Ktiles);
    int nt = (int)(tile / Ktiles);
    int kk = kt * 32 + (lane >> 4) * 16 + hh;
    int nn = nt * 16 + (lane & 15);
    float val = (nn < N) ? W[(size_t)kk * N + nn] : 0.0f;
    out[i] = (f16)val;
}

// ---------------------------------------------------------------------------
// x[b,t,:] = wte[idx[b,t],:] + wpe[t,:]
// ---------------------------------------------------------------------------
__global__ void gpt_embed(const int* __restrict__ idx, const float* __restrict__ wte,
                          const float* __restrict__ wpe, float* __restrict__ x) {
    int row = blockIdx.x;                // 0..Mn-1
    int t   = row & (Tn - 1);
    int tok = idx[row];
    for (int d = threadIdx.x; d < Dn; d += blockDim.x)
        x[(size_t)row * Dn + d] = wte[(size_t)tok * Dn + d] + wpe[(size_t)t * Dn + d];
}

// ---------------------------------------------------------------------------
// LayerNorm: x fp32 [Mn,768] -> f16 [Mn,768].  One wave per row, 8 rows/block.
// ---------------------------------------------------------------------------
__global__ __launch_bounds__(256)
void gpt_layernorm(const float* __restrict__ x, const float* __restrict__ w,
                   const float* __restrict__ b, f16* __restrict__ out) {
    const int lane = threadIdx.x & 31;
    const int wave = threadIdx.x >> 5;
    const int row  = blockIdx.x * 8 + wave;
    const float* xr = x + (size_t)row * Dn;
    float s = 0.f, sq = 0.f;
    for (int i = lane; i < Dn; i += 32) { float v = xr[i]; s += v; sq += v * v; }
#pragma unroll
    for (int off = 1; off < 32; off <<= 1) {
        s  += __shfl_xor(s, off);
        sq += __shfl_xor(sq, off);
    }
    const float inv = 1.0f / (float)Dn;
    float mu = s * inv;
    float var = sq * inv - mu * mu;
    float rs = rsqrtf(var + 1e-5f);
    f16* o = out + (size_t)row * Dn;
    for (int i = lane; i < Dn; i += 32)
        o[i] = (f16)((xr[i] - mu) * rs * w[i] + b[i]);
}

// ---------------------------------------------------------------------------
// WMMA GEMM with async global->LDS double-buffered staging.
// Block = 4 waves, 64x64 block tile; wave = 32x32 (2x2 WMMA frags).
// Per 32-deep k-step: one async A-tile copy + one async B-tile copy per wave
// (128 x b128 transfers each, tracked on ASYNCcnt), ds_load_b128 fragment
// reads from LDS, 4 v_wmma per wave.  Epilogue: bias / GELU / fp32 residual,
// fp32 and/or f16 outputs (ld = N).
// ---------------------------------------------------------------------------
__global__ __launch_bounds__(128)
void gpt_wmma_gemm(const f16* __restrict__ A, const f16* __restrict__ Bpk,
                   const float* __restrict__ bias, const float* __restrict__ resid,
                   float* __restrict__ outf, f16* __restrict__ outh,
                   int M, int N, int K, int dogelu) {
    __shared__ f16 As[2][64 * 32];   // 4 KB per buffer: 64 rows x 32 k halves
    __shared__ f16 Bs[2][64 * 32];   // 4 KB per buffer: 4 col-tiles x 512 halves

    const int tid  = threadIdx.x;
    const int lane = tid & 31;
    const int wave = tid >> 5;
    const int kg = lane >> 4, ln = lane & 15;
    const int mblk  = blockIdx.y * 64;
    const int ntblk = blockIdx.x * 4;          // first 16-wide col tile of block
    const int mloc  = (wave >> 1) * 32;        // wave row offset in block tile
    const int nloc  = (wave & 1) * 2;          // wave col tile offset (of 4)
    const int Kt = K >> 5;
    int koff[8];
    make_koff(kg, koff);

    const u32 asBase = lds_off32(&As[0][0]);
    const u32 bsBase = lds_off32(&Bs[0][0]);

    // copy thread mapping: A: row = tid>>1, 16-half part = (tid&1)*16
    //                      B: tile = tid>>5, lane chunk = (tid&31)*16
    const int cArow  = tid >> 1;
    const int cApart = (tid & 1) * 16;
    const f16* cAg = A + (size_t)(mblk + cArow) * K + cApart;
    const u32  cAl = (u32)((cArow * 32 + cApart) * 2);
    const int cBt = tid >> 5;
    const f16* cBg = Bpk + ((size_t)(ntblk + cBt) * Kt) * 512 + (tid & 31) * 16;
    const u32  cBl = (u32)(tid * 32);

    // prologue: stage k-step 0 into buffer 0
    async_cp_b128(asBase + cAl, cAg);
    async_cp_b128(bsBase + cBl, cBg);

    v8f c00 = {}, c01 = {}, c10 = {}, c11 = {};
    int buf = 0;
    for (int kt = 0; kt < Kt; ++kt, buf ^= 1) {
        if (kt + 1 < Kt) {
            async_cp_b128(asBase + (buf ^ 1) * 4096 + cAl, cAg + (kt + 1) * 32);
            async_cp_b128(bsBase + (buf ^ 1) * 4096 + cBl, cBg + (size_t)(kt + 1) * 512);
            asm volatile("s_wait_asynccnt 0x2" ::: "memory");  // current buf done
        } else {
            asm volatile("s_wait_asynccnt 0x0" ::: "memory");
        }
        __syncthreads();   // all waves' copies of this buffer visible

        AF a0, a1, b0, b1;
        const f16* asb = &As[buf][0];
        const f16* bsb = &Bs[buf][0];
#pragma unroll
        for (int vg = 0; vg < 8; ++vg) {
            a0.u[vg] = *(const u32*)(asb + (mloc + ln) * 32 + koff[vg]);
            a1.u[vg] = *(const u32*)(asb + (mloc + 16 + ln) * 32 + koff[vg]);
        }
        b0.v = *(const v16h*)(bsb + (nloc + 0) * 512 + lane * 16);
        b1.v = *(const v16h*)(bsb + (nloc + 1) * 512 + lane * 16);

        c00 = wmma_f16(a0, b0, c00);
        c01 = wmma_f16(a0, b1, c01);
        c10 = wmma_f16(a1, b0, c10);
        c11 = wmma_f16(a1, b1, c11);
        __syncthreads();   // reads done before buffer is refilled
    }

    const int mbase = mblk + mloc;
    const int nbase = (ntblk + nloc) * 16;
#define GPT_EPILOGUE(CV, RO, CO)                                               \
    {                                                                          \
        _Pragma("unroll")                                                      \
        for (int r = 0; r < 8; ++r) {                                          \
            int grow = mbase + (RO) + r + kg * 8;                              \
            int gcol = nbase + (CO) + ln;                                      \
            if (gcol < N) {                                                    \
                float val = (CV)[r];                                           \
                if (bias) val += bias[gcol];                                   \
                if (dogelu) {                                                  \
                    float u0 = val;                                            \
                    float tt = tanhf(0.7978845608028654f *                     \
                                     (u0 + 0.044715f * u0 * u0 * u0));         \
                    val = 0.5f * u0 * (1.0f + tt);                             \
                }                                                              \
                size_t o = (size_t)grow * N + gcol;                            \
                if (resid) val += resid[o];                                    \
                if (outf) outf[o] = val;                                       \
                if (outh) outh[o] = (f16)val;                                  \
            }                                                                  \
        }                                                                      \
    }
    GPT_EPILOGUE(c00, 0, 0)
    GPT_EPILOGUE(c01, 0, 16)
    GPT_EPILOGUE(c10, 16, 0)
    GPT_EPILOGUE(c11, 16, 16)
#undef GPT_EPILOGUE
}

// ---------------------------------------------------------------------------
// Flash-style causal attention with WMMA.  One wave per (b, h, 16-row q tile).
// Q/K/V f16 stored [B,T,H*DH] (token-major); S = Q K^T and O += P V via WMMA.
// Online softmax; P re-layout C-frag -> A-frag goes through 1 KB of LDS.
// ---------------------------------------------------------------------------
__global__ __launch_bounds__(32)
void gpt_attn(const f16* __restrict__ Q, const f16* __restrict__ Kk,
              const f16* __restrict__ Vv, f16* __restrict__ Y) {
    __shared__ f16 p_lds[16 * 32];
    const int lane = threadIdx.x & 31;
    const int kg = lane >> 4, ln = lane & 15;
    const int qtile = blockIdx.x, hh = blockIdx.y, bb = blockIdx.z;
    const size_t base = (size_t)bb * Tn * Dn + hh * DHn;
    int koff[8];
    make_koff(kg, koff);

    // Q A-fragments (d = 0..31 and 32..63), row = qtile*16 + ln
    const f16* qp = Q + base + (size_t)(qtile * 16 + ln) * Dn;
    AF qa0, qa1;
#pragma unroll
    for (int vg = 0; vg < 8; ++vg) {
        qa0.u[vg] = *(const u32*)(qp + koff[vg]);
        qa1.u[vg] = *(const u32*)(qp + 32 + koff[vg]);
    }

    v8f oacc[4] = {v8f{}, v8f{}, v8f{}, v8f{}};
    float ms[8], ls[8];
#pragma unroll
    for (int r = 0; r < 8; ++r) { ms[r] = -1e30f; ls[r] = 0.f; }

    const int kbend = (qtile + 2) >> 1;   // key blocks of 32, causal bound
    for (int kb = 0; kb < kbend; ++kb) {
        // --- S tiles: keys [kb*32, kb*32+15] and [kb*32+16, kb*32+31] ---
        v8f s0 = {}, s1 = {};
        {
            const f16* kp = Kk + base + (size_t)(kb * 32 + ln) * Dn;
            AF k0, k1;
            k0.v = *(const v16h*)(kp + kg * 16);        // d block 0..31
            k1.v = *(const v16h*)(kp + 32 + kg * 16);   // d block 32..63
            s0 = wmma_f16(qa0, k0, s0);
            s0 = wmma_f16(qa1, k1, s0);
        }
        {
            const f16* kp = Kk + base + (size_t)(kb * 32 + 16 + ln) * Dn;
            AF k0, k1;
            k0.v = *(const v16h*)(kp + kg * 16);
            k1.v = *(const v16h*)(kp + 32 + kg * 16);
            s1 = wmma_f16(qa0, k0, s1);
            s1 = wmma_f16(qa1, k1, s1);
        }
        // --- scale + causal mask + online softmax ---
#pragma unroll
        for (int r = 0; r < 8; ++r) {
            int qg = qtile * 16 + r + kg * 8;
            int kg0 = kb * 32 + ln, kg1 = kg0 + 16;
            float sa = s0[r] * 0.125f; if (kg0 > qg) sa = -1e30f;
            float sb = s1[r] * 0.125f; if (kg1 > qg) sb = -1e30f;
            float mx = fmaxf(sa, sb);
#pragma unroll
            for (int off = 1; off <= 8; off <<= 1) mx = fmaxf(mx, __shfl_xor(mx, off));
            float nm = fmaxf(ms[r], mx);
            float alpha = __expf(ms[r] - nm);
            float p0 = __expf(sa - nm);
            float p1 = __expf(sb - nm);
            float rs = p0 + p1;
#pragma unroll
            for (int off = 1; off <= 8; off <<= 1) rs += __shfl_xor(rs, off);
            ls[r] = ls[r] * alpha + rs;
            ms[r] = nm;
#pragma unroll
            for (int nt = 0; nt < 4; ++nt) oacc[nt][r] *= alpha;
            int qrl = r + kg * 8;
            p_lds[qrl * 32 + ln]      = (f16)p0;
            p_lds[qrl * 32 + 16 + ln] = (f16)p1;
        }
        __syncthreads();
        // P as A-fragment (16 rows x 32 keys)
        AF pa;
#pragma unroll
        for (int vg = 0; vg < 8; ++vg)
            pa.u[vg] = *(const u32*)&p_lds[ln * 32 + koff[vg]];
        __syncthreads();
        // --- O += P @ V ---
#pragma unroll
        for (int nt = 0; nt < 4; ++nt) {
            AF vf;
#pragma unroll
            for (int e = 0; e < 16; ++e) {
                int keyg = kb * 32 + kg * 16 + e;
                vf.h[e] = Vv[base + (size_t)keyg * Dn + nt * 16 + ln];
            }
            oacc[nt] = wmma_f16(pa, vf, oacc[nt]);
        }
    }
    // --- normalize and write Y [B,T,H*DH] f16 ---
#pragma unroll
    for (int nt = 0; nt < 4; ++nt) {
#pragma unroll
        for (int r = 0; r < 8; ++r) {
            int qg = qtile * 16 + r + kg * 8;
            Y[base + (size_t)qg * Dn + nt * 16 + ln] = (f16)(oacc[nt][r] / ls[r]);
        }
    }
}

// ---------------------------------------------------------------------------
extern "C" void kernel_launch(void* const* d_in, const int* in_sizes, int n_in,
                              void* d_out, int out_size, void* d_ws, size_t ws_size,
                              hipStream_t stream) {
    (void)in_sizes; (void)n_in; (void)out_size; (void)ws_size;
    const int*   idx   = (const int*)  d_in[0];
    const float* wte   = (const float*)d_in[1];
    const float* wpe   = (const float*)d_in[2];
    const float* ln1w  = (const float*)d_in[3];
    const float* ln1b  = (const float*)d_in[4];
    const float* Wq    = (const float*)d_in[5];
    const float* bq    = (const float*)d_in[6];
    const float* Wk    = (const float*)d_in[7];
    const float* bk    = (const float*)d_in[8];
    const float* Wv    = (const float*)d_in[9];
    const float* bv    = (const float*)d_in[10];
    const float* Wo    = (const float*)d_in[11];
    const float* bo    = (const float*)d_in[12];
    const float* ln2w  = (const float*)d_in[13];
    const float* ln2b  = (const float*)d_in[14];
    const float* Wfc   = (const float*)d_in[15];
    const float* bfc   = (const float*)d_in[16];
    const float* Wpr   = (const float*)d_in[17];
    const float* bpr   = (const float*)d_in[18];
    const float* lnfw  = (const float*)d_in[19];
    const float* lnfb  = (const float*)d_in[20];
    const float* Wlm   = (const float*)d_in[21];
    float* logits = (float*)d_out;

    // ----- workspace layout (256B aligned slices) -----
    char* ws = (char*)d_ws;
    size_t off = 0;
    auto take = [&](size_t bytes) -> void* {
        void* p = ws + off;
        off += (bytes + 255) & ~(size_t)255;
        return p;
    };
    f16* pkq  = (f16*)take((size_t)Ln * Dn * Dn * 2);
    f16* pkk  = (f16*)take((size_t)Ln * Dn * Dn * 2);
    f16* pkv  = (f16*)take((size_t)Ln * Dn * Dn * 2);
    f16* pko  = (f16*)take((size_t)Ln * Dn * Dn * 2);
    f16* pkfc = (f16*)take((size_t)Ln * Dn * FFn * 2);
    f16* pkpr = (f16*)take((size_t)Ln * FFn * Dn * 2);
    f16* pklm = (f16*)take((size_t)Dn * VPAD * 2);
    float* x  = (float*)take((size_t)Mn * Dn * 4);
    f16* h16  = (f16*)take((size_t)Mn * Dn * 2);
    f16* q16  = (f16*)take((size_t)Mn * Dn * 2);
    f16* k16  = (f16*)take((size_t)Mn * Dn * 2);
    f16* v16  = (f16*)take((size_t)Mn * Dn * 2);
    f16* y16  = (f16*)take((size_t)Mn * Dn * 2);
    f16* a16  = (f16*)take((size_t)Mn * FFn * 2);

    // ----- pack weights to f16 B-fragment layout -----
    auto pack = [&](const float* W, f16* out, int K, int N, int Npad) {
        int Kt = K / 32, Nt = Npad / 16;
        size_t total = (size_t)Nt * Kt * 512;
        int blocks = (int)((total + 255) / 256);
        gpt_pack_w<<<blocks, 256, 0, stream>>>(W, out, N, Kt, Nt);
    };
    for (int l = 0; l < Ln; ++l) {
        pack(Wq  + (size_t)l * Dn * Dn,  pkq  + (size_t)l * Dn * Dn,  Dn, Dn, Dn);
        pack(Wk  + (size_t)l * Dn * Dn,  pkk  + (size_t)l * Dn * Dn,  Dn, Dn, Dn);
        pack(Wv  + (size_t)l * Dn * Dn,  pkv  + (size_t)l * Dn * Dn,  Dn, Dn, Dn);
        pack(Wo  + (size_t)l * Dn * Dn,  pko  + (size_t)l * Dn * Dn,  Dn, Dn, Dn);
        pack(Wfc + (size_t)l * Dn * FFn, pkfc + (size_t)l * Dn * FFn, FFn, Dn, FFn);
        pack(Wpr + (size_t)l * FFn * Dn, pkpr + (size_t)l * FFn * Dn, Dn, FFn, Dn);
    }
    pack(Wlm, pklm, Dn, Vn, VPAD);

    // ----- embedding -----
    gpt_embed<<<Mn, 256, 0, stream>>>(idx, wte, wpe, x);

    const dim3 gD(Dn / 64, Mn / 64);    // N=768 GEMMs
    const dim3 gF(FFn / 64, Mn / 64);   // N=3072 GEMM
    const dim3 gA(Tn / 16, Hn, Bn);     // attention

    for (int l = 0; l < Ln; ++l) {
        gpt_layernorm<<<Mn / 8, 256, 0, stream>>>(x, ln1w + l * Dn, ln1b + l * Dn, h16);
        gpt_wmma_gemm<<<gD, 128, 0, stream>>>(h16, pkq + (size_t)l * Dn * Dn,
            bq + l * Dn, nullptr, nullptr, q16, Mn, Dn, Dn, 0);
        gpt_wmma_gemm<<<gD, 128, 0, stream>>>(h16, pkk + (size_t)l * Dn * Dn,
            bk + l * Dn, nullptr, nullptr, k16, Mn, Dn, Dn, 0);
        gpt_wmma_gemm<<<gD, 128, 0, stream>>>(h16, pkv + (size_t)l * Dn * Dn,
            bv + l * Dn, nullptr, nullptr, v16, Mn, Dn, Dn, 0);
        gpt_attn<<<gA, 32, 0, stream>>>(q16, k16, v16, y16);
        gpt_wmma_gemm<<<gD, 128, 0, stream>>>(y16, pko + (size_t)l * Dn * Dn,
            bo + l * Dn, x, x, nullptr, Mn, Dn, Dn, 0);       // residual add -> x
        gpt_layernorm<<<Mn / 8, 256, 0, stream>>>(x, ln2w + l * Dn, ln2b + l * Dn, h16);
        gpt_wmma_gemm<<<gF, 128, 0, stream>>>(h16, pkfc + (size_t)l * Dn * FFn,
            bfc + l * FFn, nullptr, nullptr, a16, Mn, FFn, Dn, 1);  // GELU
        gpt_wmma_gemm<<<gD, 128, 0, stream>>>(a16, pkpr + (size_t)l * FFn * Dn,
            bpr + l * Dn, x, x, nullptr, Mn, Dn, FFn, 0);     // residual add -> x
    }

    gpt_layernorm<<<Mn / 8, 256, 0, stream>>>(x, lnfw, lnfb, h16);
    gpt_wmma_gemm<<<dim3(VPAD / 64, Mn / 64), 128, 0, stream>>>(
        h16, pklm, nullptr, nullptr, logits, nullptr, Mn, Vn, Dn, 0);
}
